// TransformerLayer_75118978007234
// MI455X (gfx1250) — matrix operand order repaired and checked by reading, not compiled, not run on hardware
//
#include <hip/hip_runtime.h>

// ---------------------------------------------------------------------------
// MI455X / gfx1250 transformer layer: f16 WMMA GEMMs + flash attention.
// B=16, S=1024, EMB=768, HEADS=12, HEAD_DIM=64, MLP=3072.
// Weights are pre-transposed/converted once (fp32 [K][N] -> f16 [N][K]) so
// every GEMM operand is staged with async global->LDS b128 copies and every
// fragment load is a ds_load_b128 pair feeding v_wmma_f32_16x16x32_f16.
// ---------------------------------------------------------------------------

typedef __attribute__((ext_vector_type(16))) _Float16 v16h;
typedef __attribute__((ext_vector_type(8)))  float    v8f;

#define EMB      768
#define HEADS    12
#define HEAD_DIM 64
#define MLP_DIM  3072
#define BATCH    16
#define SEQ      1024
#define MTOT     (BATCH * SEQ)   // 16384 rows

__device__ __forceinline__ v8f v8f_zero() {
    v8f z;
#pragma unroll
    for (int i = 0; i < 8; ++i) z[i] = 0.0f;
    return z;
}

__device__ __forceinline__ v8f wmma_f16(v16h a, v16h b, v8f c) {
    return __builtin_amdgcn_wmma_f32_16x16x32_f16(
        false, a, false, b, (short)0, c, false, false);
}

// ---- CDNA5 async global->LDS copy (tracked by ASYNCcnt) -------------------
__device__ __forceinline__ void async_copy_b128(void* lds_dst, const void* gsrc) {
    // LDS aperture: low 32 bits of the flat address are the LDS byte offset.
    unsigned int loff = (unsigned int)(unsigned long long)lds_dst;
    asm volatile("global_load_async_to_lds_b128 %0, %1, off"
                 :: "v"(loff), "v"(gsrc) : "memory");
}
__device__ __forceinline__ void wait_async0() {
    asm volatile("s_wait_asynccnt 0" ::: "memory");
}

// ---- fragment loaders (all b128) ------------------------------------------
// A-fragment 16x32 f16 (ISA 7.12.2): per lane two contiguous 8-half runs.
__device__ __forceinline__ v16h load_frag_a(const _Float16* base, int ld,
                                            int m0, int k0, int lane) {
    const int half = (lane >> 4) & 1;
    const int mr   = lane & 15;
    const _Float16* p = base + (m0 + mr) * ld + k0 + 8 * half;
    union { v16h v; float4 f[2]; } u;
    u.f[0] = *reinterpret_cast<const float4*>(p);        // k = 8h .. 8h+7
    u.f[1] = *reinterpret_cast<const float4*>(p + 16);   // k = 16+8h .. 16+8h+7
    return u.v;
}

// B-fragment 32x16 f16 from K-contiguous storage base[n*ld + k]:
// lane's 16 halves are k = k0 + 8*half + i and k0 + 16 + 8*half + i (i<8).
__device__ __forceinline__ v16h load_frag_b_t(const _Float16* base, int ld,
                                              int n0, int k0, int lane) {
    const int half = (lane >> 4) & 1;
    const int n    = lane & 15;
    const _Float16* p = base + (n0 + n) * ld + k0 + 8 * half;
    union { v16h v; float4 f[2]; } u;
    u.f[0] = *reinterpret_cast<const float4*>(p);
    u.f[1] = *reinterpret_cast<const float4*>(p + 16);
    return u.v;
}

// ---------------------------------------------------------------------------
// One-time weight prep: W fp32 [K][N] -> Wt f16 [N][K] (K-contiguous).
// 64x64 tiles through padded LDS; coalesced reads and b128 writes.
// ---------------------------------------------------------------------------
__global__ void __launch_bounds__(256)
transpose_cvt(const float* __restrict__ in, _Float16* __restrict__ out,
              int K, int N) {
    __shared__ __align__(16) _Float16 T[64 * 80];   // [n][k], padded stride
    const int tid = threadIdx.x;
    const int n0  = blockIdx.x * 64;
    const int k0  = blockIdx.y * 64;

#pragma unroll
    for (int it = 0; it < 4; ++it) {
        const int c  = tid + 256 * it;     // 0..1023
        const int r  = c >> 4;             // k row 0..63
        const int no = (c & 15) * 4;
        const float4 f = *reinterpret_cast<const float4*>(
            &in[(size_t)(k0 + r) * N + n0 + no]);
        T[(no + 0) * 80 + r] = (_Float16)f.x;
        T[(no + 1) * 80 + r] = (_Float16)f.y;
        T[(no + 2) * 80 + r] = (_Float16)f.z;
        T[(no + 3) * 80 + r] = (_Float16)f.w;
    }
    __syncthreads();
#pragma unroll
    for (int it = 0; it < 2; ++it) {
        const int c  = tid + 256 * it;     // 0..511
        const int n  = c >> 3;             // 0..63
        const int kq = (c & 7) * 8;        // 0..56
        *reinterpret_cast<float4*>(&out[(size_t)(n0 + n) * K + k0 + kq]) =
            *reinterpret_cast<const float4*>(&T[n * 80 + kq]);
    }
}

// ---------------------------------------------------------------------------
// LayerNorm(fp32 in) -> f16 out. One 256-thread block per 768-wide row.
// ---------------------------------------------------------------------------
__global__ void __launch_bounds__(256)
layernorm_f16(const float* __restrict__ x, const float* __restrict__ g,
              const float* __restrict__ bta, _Float16* __restrict__ out) {
    __shared__ float red[256];
    const int row = blockIdx.x;
    const int tid = threadIdx.x;
    const float* xr = x + (size_t)row * EMB;

    float vals[3];
    float s = 0.0f;
#pragma unroll
    for (int j = 0; j < 3; ++j) { vals[j] = xr[tid + 256 * j]; s += vals[j]; }
    red[tid] = s;
    __syncthreads();
    for (int st = 128; st > 0; st >>= 1) {
        if (tid < st) red[tid] += red[tid + st];
        __syncthreads();
    }
    const float mu = red[0] * (1.0f / EMB);
    __syncthreads();

    float s2 = 0.0f;
#pragma unroll
    for (int j = 0; j < 3; ++j) { const float d = vals[j] - mu; s2 += d * d; }
    red[tid] = s2;
    __syncthreads();
    for (int st = 128; st > 0; st >>= 1) {
        if (tid < st) red[tid] += red[tid + st];
        __syncthreads();
    }
    const float rs = rsqrtf(red[0] * (1.0f / EMB) + 1e-6f);
#pragma unroll
    for (int j = 0; j < 3; ++j) {
        const int col = tid + 256 * j;
        out[(size_t)row * EMB + col] =
            (_Float16)((vals[j] - mu) * rs * g[col] + bta[col]);
    }
}

// ---------------------------------------------------------------------------
// WMMA GEMM: out[M,N] = epilogue(A_f16[M,K] @ Wt_f16[N][K] + bias (+resid)).
// 256 threads = 8 waves; block tile 128x128, K step 32. Wave w owns rows
// 16w..16w+15 across 128 cols (8 accumulators = 64 WMMAs per K step/block).
// Both operands staged by async global->LDS b128 (pure copies, no converts).
// ---------------------------------------------------------------------------
template <typename OutT, int ACT, bool RESID>
__global__ void __launch_bounds__(256)
gemm_wmma(const _Float16* __restrict__ A, const _Float16* __restrict__ Wt,
          const float* __restrict__ bias, const float* __restrict__ resid,
          OutT* __restrict__ out, int M, int N, int K) {
    __shared__ __align__(16) _Float16 As[128 * 32];   // 8 KB  [m][k]
    __shared__ __align__(16) _Float16 Bt[128 * 32];   // 8 KB  [n][k]

    const int tid    = threadIdx.x;
    const int lane   = tid & 31;
    const int w      = tid >> 5;
    const int half   = (lane >> 4) & 1;
    const int m_base = blockIdx.y * 128;
    const int n_base = blockIdx.x * 128;

    v8f acc[8];
#pragma unroll
    for (int nb = 0; nb < 8; ++nb) acc[nb] = v8f_zero();

    for (int k0 = 0; k0 < K; k0 += 32) {
        if (k0 + 32 < K) {  // speculative prefetch of next Wt slice into L2
            __builtin_prefetch(&Wt[(size_t)(n_base + (tid >> 1)) * K + k0 + 32 +
                                   (tid & 1) * 16], 0, 1);
        }
        // A tile 128x32 halves: async copy.
#pragma unroll
        for (int it = 0; it < 2; ++it) {
            const int c  = tid + 256 * it;     // 0..511
            const int r  = c >> 2;             // 0..127
            const int ko = (c & 3) * 8;        // 0,8,16,24
            async_copy_b128(&As[r * 32 + ko],
                            &A[(size_t)(m_base + r) * K + k0 + ko]);
        }
        // B tile 128x32 halves (already K-contiguous in Wt): async copy.
#pragma unroll
        for (int it = 0; it < 2; ++it) {
            const int c  = tid + 256 * it;     // 0..511
            const int n  = c >> 2;             // 0..127
            const int kq = (c & 3) * 8;        // 0,8,16,24
            async_copy_b128(&Bt[n * 32 + kq],
                            &Wt[(size_t)(n_base + n) * K + k0 + kq]);
        }
        wait_async0();
        __syncthreads();

        const v16h afrag = load_frag_a(As, 32, 16 * w, 0, lane);
#pragma unroll
        for (int nb = 0; nb < 8; ++nb) {
            const v16h bfrag = load_frag_b_t(Bt, 32, 16 * nb, 0, lane);
            acc[nb] = wmma_f16(afrag, bfrag, acc[nb]);
        }
        __syncthreads();
    }

    // Epilogue. C layout: VGPR j -> row j + 8*half, col = lane&15.
#pragma unroll
    for (int nb = 0; nb < 8; ++nb) {
        const int col = n_base + 16 * nb + (lane & 15);
        const float bv = bias[col];
#pragma unroll
        for (int j = 0; j < 8; ++j) {
            const int row = m_base + 16 * w + j + 8 * half;
            float v = acc[nb][j] + bv;
            if (ACT == 1) v = 0.5f * v * (1.0f + erff(v * 0.70710678118654752f));
            if (RESID)    v += resid[(size_t)row * N + col];
            out[(size_t)row * N + col] = (OutT)v;
        }
    }
}

// ---------------------------------------------------------------------------
// Flash attention: 128 threads (4 waves) per (64 q-rows, head, batch).
// K stays [t][d] (K-contiguous for QK^T); V transposed once per tile into
// Vt[d][t] so P*V fragments are b128 too. Online softmax state lives in the
// WMMA C-layout lanes.
// ---------------------------------------------------------------------------
__global__ void __launch_bounds__(128)
attention_wmma(const _Float16* __restrict__ Qg, const _Float16* __restrict__ Kg,
               const _Float16* __restrict__ Vg, _Float16* __restrict__ Ctx) {
    __shared__ __align__(16) _Float16 Qs[64 * 64];   // [s][d]
    __shared__ __align__(16) _Float16 Ks[64 * 64];   // [t][d]
    __shared__ __align__(16) _Float16 Vs[64 * 64];   // [t][d] staging
    __shared__ __align__(16) _Float16 Vt[64 * 64];   // [d][t] transposed
    __shared__ __align__(16) _Float16 Ps[64 * 64];   // probs [s][t]

    const int tid  = threadIdx.x;
    const int lane = tid & 31;
    const int w    = tid >> 5;
    const int half = (lane >> 4) & 1;
    const int s0   = blockIdx.x * 64;
    const int h    = blockIdx.y;
    const int b    = blockIdx.z;
    const float scale = 0.125f;  // 1/sqrt(64)

#pragma unroll
    for (int it = 0; it < 4; ++it) {
        const int c    = tid + 128 * it;   // 0..511
        const int r    = c >> 3;           // 0..63
        const int doff = (c & 7) * 8;      // 0..56
        async_copy_b128(&Qs[r * 64 + doff],
                        &Qg[((size_t)(b * SEQ + s0 + r)) * EMB +
                            h * HEAD_DIM + doff]);
    }

    float mrow[8], lrow[8];
    v8f oacc[4];
#pragma unroll
    for (int j = 0; j < 8; ++j) { mrow[j] = -1e30f; lrow[j] = 0.0f; }
#pragma unroll
    for (int db = 0; db < 4; ++db) oacc[db] = v8f_zero();

    wait_async0();
    __syncthreads();

    const v16h qf0 = load_frag_a(Qs, 64, 16 * w, 0, lane);
    const v16h qf1 = load_frag_a(Qs, 64, 16 * w, 32, lane);

    for (int t0 = 0; t0 < SEQ; t0 += 64) {
        __syncthreads();  // protect Ks/Vs/Vt reuse across iterations
#pragma unroll
        for (int it = 0; it < 4; ++it) {
            const int c    = tid + 128 * it;
            const int r    = c >> 3;
            const int doff = (c & 7) * 8;
            const size_t gsrc =
                ((size_t)(b * SEQ + t0 + r)) * EMB + h * HEAD_DIM + doff;
            async_copy_b128(&Ks[r * 64 + doff], &Kg[gsrc]);
            async_copy_b128(&Vs[r * 64 + doff], &Vg[gsrc]);
        }
        wait_async0();
        __syncthreads();

        // Transpose V tile: Vs[t][d] -> Vt[d][t]  (b128 reads, b16 scatters).
#pragma unroll
        for (int it = 0; it < 4; ++it) {
            const int c  = tid + 128 * it;   // 0..511
            const int r  = c & 63;           // t
            const int dq = (c >> 6) * 8;     // d quad base
            union { float4 f; _Float16 hh[8]; } u;
            u.f = *reinterpret_cast<const float4*>(&Vs[r * 64 + dq]);
#pragma unroll
            for (int j = 0; j < 8; ++j) Vt[(dq + j) * 64 + r] = u.hh[j];
        }
        __syncthreads();

        // Scores: S = scale * Q K^T.
        v8f sacc[4];
#pragma unroll
        for (int t = 0; t < 4; ++t) {
            v8f s = v8f_zero();
            s = wmma_f16(qf0, load_frag_b_t(Ks, 64, 16 * t, 0,  lane), s);
            s = wmma_f16(qf1, load_frag_b_t(Ks, 64, 16 * t, 32, lane), s);
#pragma unroll
            for (int j = 0; j < 8; ++j) s[j] *= scale;
            sacc[t] = s;
        }

        // Online softmax in registers (row j + 8*half owned by this lane).
#pragma unroll
        for (int j = 0; j < 8; ++j) {
            float rm = sacc[0][j];
#pragma unroll
            for (int t = 1; t < 4; ++t) rm = fmaxf(rm, sacc[t][j]);
            for (int msk = 1; msk < 16; msk <<= 1)
                rm = fmaxf(rm, __shfl_xor(rm, msk, 16));
            const float mnew  = fmaxf(mrow[j], rm);
            const float alpha = __expf(mrow[j] - mnew);
            mrow[j] = mnew;

            float rs = 0.0f;
#pragma unroll
            for (int t = 0; t < 4; ++t) {
                const float p = __expf(sacc[t][j] - mnew);
                Ps[(16 * w + j + 8 * half) * 64 + 16 * t + (lane & 15)] =
                    (_Float16)p;
                rs += p;
            }
            for (int msk = 1; msk < 16; msk <<= 1)
                rs += __shfl_xor(rs, msk, 16);
            lrow[j] = lrow[j] * alpha + rs;
#pragma unroll
            for (int db = 0; db < 4; ++db) oacc[db][j] *= alpha;
        }

        // O += P V  (P from per-wave LDS strip; V from transposed tile).
#pragma unroll
        for (int tb = 0; tb < 2; ++tb) {
            const v16h pf = load_frag_a(Ps, 64, 16 * w, 32 * tb, lane);
#pragma unroll
            for (int db = 0; db < 4; ++db) {
                const v16h vf = load_frag_b_t(Vt, 64, 16 * db, 32 * tb, lane);
                oacc[db] = wmma_f16(pf, vf, oacc[db]);
            }
        }
    }

    // Normalize and write ctx as f16 [B,S,H*D].
#pragma unroll
    for (int db = 0; db < 4; ++db) {
        const int d = 16 * db + (lane & 15);
#pragma unroll
        for (int j = 0; j < 8; ++j) {
            const int row = s0 + 16 * w + j + 8 * half;
            const float v = oacc[db][j] / lrow[j];
            Ctx[((size_t)(b * SEQ + row)) * EMB + h * HEAD_DIM + d] =
                (_Float16)v;
        }
    }
}

// ---------------------------------------------------------------------------
// Host-side orchestration.
// ---------------------------------------------------------------------------
extern "C" void kernel_launch(void* const* d_in, const int* in_sizes, int n_in,
                              void* d_out, int out_size, void* d_ws,
                              size_t ws_size, hipStream_t stream) {
    (void)in_sizes; (void)n_in; (void)out_size; (void)ws_size;

    const float* x     = (const float*)d_in[0];
    const float* ln1_g = (const float*)d_in[1];
    const float* ln1_b = (const float*)d_in[2];
    const float* Wq    = (const float*)d_in[3];
    const float* bq    = (const float*)d_in[4];
    const float* Wk    = (const float*)d_in[5];
    const float* bk    = (const float*)d_in[6];
    const float* Wv    = (const float*)d_in[7];
    const float* bv    = (const float*)d_in[8];
    const float* Wo    = (const float*)d_in[9];
    const float* bo    = (const float*)d_in[10];
    const float* ln2_g = (const float*)d_in[11];
    const float* ln2_b = (const float*)d_in[12];
    const float* W1    = (const float*)d_in[13];
    const float* b1    = (const float*)d_in[14];
    const float* W2    = (const float*)d_in[15];
    const float* b2    = (const float*)d_in[16];

    char* ws = (char*)d_ws;
    const size_t SZ_H  = (size_t)MTOT * EMB * sizeof(_Float16);       // 25.2 MB
    const size_t SZ_X1 = (size_t)MTOT * EMB * sizeof(float);          // 50.3 MB
    const size_t SZ_M  = (size_t)MTOT * MLP_DIM * sizeof(_Float16);   // 100.7 MB
    const size_t SZ_WE = (size_t)EMB * EMB * sizeof(_Float16);        // 1.18 MB
    const size_t SZ_W1 = (size_t)EMB * MLP_DIM * sizeof(_Float16);    // 4.72 MB

    _Float16* h    = (_Float16*)(ws);                    // LN1 out / reused as ctx
    _Float16* q    = (_Float16*)(ws + SZ_H);             // reused as h2 (LN2 out)
    _Float16* k    = (_Float16*)(ws + 2 * SZ_H);
    _Float16* v    = (_Float16*)(ws + 3 * SZ_H);
    float*    x1   = (float*)   (ws + 4 * SZ_H);
    _Float16* mbuf = (_Float16*)(ws + 4 * SZ_H + SZ_X1);
    char*     wp   = ws + 4 * SZ_H + SZ_X1 + SZ_M;
    _Float16* WqT  = (_Float16*)(wp);
    _Float16* WkT  = (_Float16*)(wp + SZ_WE);
    _Float16* WvT  = (_Float16*)(wp + 2 * SZ_WE);
    _Float16* WoT  = (_Float16*)(wp + 3 * SZ_WE);
    _Float16* W1T  = (_Float16*)(wp + 4 * SZ_WE);
    _Float16* W2T  = (_Float16*)(wp + 4 * SZ_WE + SZ_W1);
    _Float16* ctx  = h;
    _Float16* h2   = q;

    const dim3 blk256(256);
    const dim3 grdLN(MTOT);
    const dim3 grdG_E(EMB / 128, MTOT / 128);      // (6, 128)
    const dim3 grdG_M(MLP_DIM / 128, MTOT / 128);  // (24, 128)
    const dim3 grdAtt(SEQ / 64, HEADS, BATCH);     // (16, 12, 16)
    const dim3 grdT_E(EMB / 64, EMB / 64);         // (12, 12)
    const dim3 grdT_1(MLP_DIM / 64, EMB / 64);     // (48, 12)
    const dim3 grdT_2(EMB / 64, MLP_DIM / 64);     // (12, 48)

    // 0) One-time weight transpose + fp32->f16 convert.
    transpose_cvt<<<grdT_E, blk256, 0, stream>>>(Wq, WqT, EMB, EMB);
    transpose_cvt<<<grdT_E, blk256, 0, stream>>>(Wk, WkT, EMB, EMB);
    transpose_cvt<<<grdT_E, blk256, 0, stream>>>(Wv, WvT, EMB, EMB);
    transpose_cvt<<<grdT_E, blk256, 0, stream>>>(Wo, WoT, EMB, EMB);
    transpose_cvt<<<grdT_1, blk256, 0, stream>>>(W1, W1T, EMB, MLP_DIM);
    transpose_cvt<<<grdT_2, blk256, 0, stream>>>(W2, W2T, MLP_DIM, EMB);

    // 1) LN1 -> h (f16)
    layernorm_f16<<<grdLN, blk256, 0, stream>>>(x, ln1_g, ln1_b, h);

    // 2) QKV projections (f16 out, [B,S,H*D])
    gemm_wmma<_Float16, 0, false><<<grdG_E, blk256, 0, stream>>>(
        h, WqT, bq, nullptr, q, MTOT, EMB, EMB);
    gemm_wmma<_Float16, 0, false><<<grdG_E, blk256, 0, stream>>>(
        h, WkT, bk, nullptr, k, MTOT, EMB, EMB);
    gemm_wmma<_Float16, 0, false><<<grdG_E, blk256, 0, stream>>>(
        h, WvT, bv, nullptr, v, MTOT, EMB, EMB);

    // 3) Flash attention -> ctx (f16)
    attention_wmma<<<grdAtt, dim3(128), 0, stream>>>(q, k, v, ctx);

    // 4) Out projection + residual -> x1 (fp32)
    gemm_wmma<float, 0, true><<<grdG_E, blk256, 0, stream>>>(
        ctx, WoT, bo, x, x1, MTOT, EMB, EMB);

    // 5) LN2 -> h2 (f16)
    layernorm_f16<<<grdLN, blk256, 0, stream>>>(x1, ln2_g, ln2_b, h2);

    // 6) MLP: GEMM1 + exact GELU (f16), GEMM2 + residual -> d_out (fp32)
    gemm_wmma<_Float16, 1, false><<<grdG_M, blk256, 0, stream>>>(
        h2, W1T, b1, nullptr, mbuf, MTOT, MLP_DIM, EMB);
    gemm_wmma<float, 0, true><<<grdG_E, blk256, 0, stream>>>(
        mbuf, W2T, b2, x1, (float*)d_out, MTOT, EMB, MLP_DIM);
}
